// HyperGraphHead_83030307766610
// MI455X (gfx1250) — compile-verified, matrix operand order checked by reading, not run on hardware
//
#include <hip/hip_runtime.h>
#include <hip/hip_bf16.h>
#include <math.h>

#define BB 8
#define NN 2048
#define HH 512

typedef __attribute__((ext_vector_type(8)))  __bf16 v8bf;
typedef __attribute__((ext_vector_type(16))) __bf16 v16bf;
typedef __attribute__((ext_vector_type(8)))  float  v8f;

__device__ __forceinline__ v16bf load_frag16(const __bf16* plo, const __bf16* phi) {
  union { v16bf v; v8bf h[2]; } u;
  u.h[0] = *(const v8bf*)plo;
  u.h[1] = *(const v8bf*)phi;
  return u.v;
}

// ---------------- normalize: ebf = bf16(embed), qbf = bf16(embed / max(||embed||,1e-12))
__global__ __launch_bounds__(256) void normalize_kernel(const float* __restrict__ embed,
                                                        __bf16* __restrict__ ebf,
                                                        __bf16* __restrict__ qbf) {
  int row  = blockIdx.x * 8 + (threadIdx.x >> 5);
  int lane = threadIdx.x & 31;
  const float* e = embed + (size_t)row * HH;
  float ss = 0.f;
  for (int i = lane; i < HH; i += 32) { float v = e[i]; ss += v * v; }
  #pragma unroll
  for (int o = 16; o > 0; o >>= 1) ss += __shfl_xor(ss, o);
  float inv = 1.0f / fmaxf(sqrtf(ss), 1e-12f);
  for (int i = lane; i < HH; i += 32) {
    float v = e[i];
    ebf[(size_t)row * HH + i] = (__bf16)v;
    qbf[(size_t)row * HH + i] = (__bf16)(v * inv);
  }
}

// ---------------- convert + transpose fp32 W[K][Nn] -> bf16 Wt[Nn][K]
__global__ void prep_wt_kernel(const float* __restrict__ W, __bf16* __restrict__ Wt,
                               int K, int Nn) {
  int i = blockIdx.x * blockDim.x + threadIdx.x;
  if (i >= K * Nn) return;
  int n = i / K, k = i % K;
  Wt[i] = (__bf16)W[(size_t)k * Nn + n];
}

// ---------------- fused correlation + row argmax: idx[b,n] = argmax_m q[n].q[m]; cnt = in-degree
__global__ __launch_bounds__(128) void corr_argmax_kernel(const __bf16* __restrict__ q,
                                                          int* __restrict__ idx,
                                                          int* __restrict__ cnt) {
  const int TN = NN / 16;               // 128 column tiles
  int b    = blockIdx.x / TN;
  int m0   = (blockIdx.x % TN) * 16;
  int wave = threadIdx.x >> 5, lane = threadIdx.x & 31;
  int r    = lane & 15, half = lane >> 4, koff = half * 8;
  const __bf16* qb   = q + (size_t)b * NN * HH;
  const __bf16* arow = qb + (size_t)(m0 + r) * HH;

  float bestv[8]; int bestc[8];
  #pragma unroll
  for (int v = 0; v < 8; ++v) { bestv[v] = -1e30f; bestc[v] = 0; }

  for (int jt = wave; jt < TN; jt += 4) {
    const __bf16* brow = qb + (size_t)(jt * 16 + r) * HH;
    v8f acc = {};
    #pragma unroll
    for (int kt = 0; kt < 16; ++kt) {
      int k0 = kt * 32 + koff;
      v16bf aF = load_frag16(arow + k0, arow + k0 + 16);
      v16bf bF = load_frag16(brow + k0, brow + k0 + 16);
      acc = __builtin_amdgcn_wmma_f32_16x16x32_bf16(false, aF, false, bF,
                                                    (short)0, acc, false, false);
    }
    int col = jt * 16 + r;
    #pragma unroll
    for (int v = 0; v < 8; ++v) {
      float s = acc[v];
      if (s > bestv[v]) { bestv[v] = s; bestc[v] = col; }
    }
  }

  __shared__ float sv[4][16][16];
  __shared__ int   sc[4][16][16];
  #pragma unroll
  for (int v = 0; v < 8; ++v) {
    int m = v + 8 * half;
    sv[wave][m][r] = bestv[v];
    sc[wave][m][r] = bestc[v];
  }
  __syncthreads();
  if (threadIdx.x < 16) {
    int m = threadIdx.x;
    float bv = -1e30f; int bc = 0;
    for (int w = 0; w < 4; ++w)
      for (int l = 0; l < 16; ++l) {
        float v = sv[w][m][l]; int c = sc[w][m][l];
        if (v > bv || (v == bv && c < bc)) { bv = v; bc = c; }   // top_k tie: lowest index
      }
    idx[(size_t)b * NN + m0 + m] = bc;
    atomicAdd(&cnt[(size_t)b * NN + bc], 1);
  }
}

// ---------------- diagonal of G
__global__ void gscale_kernel(const int* __restrict__ idx, const int* __restrict__ cnt,
                              float* __restrict__ g) {
  int i = blockIdx.x * blockDim.x + threadIdx.x;
  if (i >= BB * NN) return;
  int n = i & (NN - 1);
  float val = 0.f;
  if (idx[i] == n) {
    float de = (float)cnt[i];
    val = (de > 0.f) ? (1.0f / 1.0001f) * (1.0f / (de + 1e-4f)) : 0.f;
  }
  g[i] = val;
}

// ---------------- GEMM: out = maybe_relu( g[m] * (A @ W + bias) ), bf16 out
__global__ __launch_bounds__(256) void gemm_bf16_kernel(const __bf16* __restrict__ A,
                                                        const __bf16* __restrict__ Wt,
                                                        const float* __restrict__ bias,
                                                        const float* __restrict__ g,
                                                        __bf16* __restrict__ out,
                                                        int Nn, int K, int relu) {
  int wave = threadIdx.x >> 5, lane = threadIdx.x & 31;
  int r = lane & 15, half = lane >> 4, koff = half * 8;
  int tiles_n = Nn >> 4;
  int tile = blockIdx.x * 8 + wave;
  int m0 = (tile / tiles_n) * 16, n0 = (tile % tiles_n) * 16;
  const __bf16* arow = A  + (size_t)(m0 + r) * K;
  const __bf16* brow = Wt + (size_t)(n0 + r) * K;
  v8f acc = {};
  for (int kt = 0; kt < (K >> 5); ++kt) {
    int k0 = kt * 32 + koff;
    v16bf aF = load_frag16(arow + k0, arow + k0 + 16);
    v16bf bF = load_frag16(brow + k0, brow + k0 + 16);
    acc = __builtin_amdgcn_wmma_f32_16x16x32_bf16(false, aF, false, bF,
                                                  (short)0, acc, false, false);
  }
  int nc = n0 + r;
  float bv = bias[nc];
  #pragma unroll
  for (int v = 0; v < 8; ++v) {
    int m = m0 + v + 8 * half;
    float y = (acc[v] + bv) * g[m];
    if (relu) y = fmaxf(y, 0.f);
    out[(size_t)m * Nn + nc] = (__bf16)y;
  }
}

// ---------------- fused concat-GEMM (K=1024 via two K=512 passes) + bias + LayerNorm
__global__ __launch_bounds__(256) void fc_ln_kernel(const __bf16* __restrict__ ebf,
                                                    const __bf16* __restrict__ x2bf,
                                                    const __bf16* __restrict__ Wfct,
                                                    const float* __restrict__ bfc,
                                                    const float* __restrict__ ln_g,
                                                    const float* __restrict__ ln_b,
                                                    float* __restrict__ out) {
  const int LSTR = 516;                      // pad: 516 % 64 != 0 -> no bank conflicts
  __shared__ float yt[16 * LSTR];
  int m0   = blockIdx.x * 16;
  int wave = threadIdx.x >> 5, lane = threadIdx.x & 31;
  int r = lane & 15, half = lane >> 4, koff = half * 8;

  const __bf16* arow0 = ebf  + (size_t)(m0 + r) * HH;
  const __bf16* arow1 = x2bf + (size_t)(m0 + r) * HH;
  for (int t = 0; t < 4; ++t) {
    int n0 = (wave * 4 + t) * 16;
    const __bf16* brow = Wfct + (size_t)(n0 + r) * (2 * HH);
    v8f acc = {};
    #pragma unroll
    for (int kt = 0; kt < 16; ++kt) {        // embed half of concat
      int k0 = kt * 32 + koff;
      v16bf aF = load_frag16(arow0 + k0, arow0 + k0 + 16);
      v16bf bF = load_frag16(brow + k0, brow + k0 + 16);
      acc = __builtin_amdgcn_wmma_f32_16x16x32_bf16(false, aF, false, bF,
                                                    (short)0, acc, false, false);
    }
    #pragma unroll
    for (int kt = 0; kt < 16; ++kt) {        // x2 half of concat
      int k0 = kt * 32 + koff;
      v16bf aF = load_frag16(arow1 + k0, arow1 + k0 + 16);
      v16bf bF = load_frag16(brow + HH + k0, brow + HH + k0 + 16);
      acc = __builtin_amdgcn_wmma_f32_16x16x32_bf16(false, aF, false, bF,
                                                    (short)0, acc, false, false);
    }
    int nc = n0 + r;
    float bv = bfc[nc];
    #pragma unroll
    for (int v = 0; v < 8; ++v)
      yt[(v + 8 * half) * LSTR + nc] = acc[v] + bv;
  }
  __syncthreads();

  int row = threadIdx.x >> 4;                // 16 threads per row
  int c0  = (threadIdx.x & 15) * 32;
  float s = 0.f, ssq = 0.f;
  for (int j = 0; j < 32; ++j) {
    float v = yt[row * LSTR + c0 + j];
    s += v; ssq += v * v;
  }
  #pragma unroll
  for (int o = 8; o > 0; o >>= 1) { s += __shfl_xor(s, o); ssq += __shfl_xor(ssq, o); }
  float mu  = s * (1.f / HH);
  float var = ssq * (1.f / HH) - mu * mu;
  float rs  = rsqrtf(var + 1e-5f);
  float* orow = out + (size_t)(m0 + row) * HH;
  for (int j = 0; j < 32; ++j) {
    int c = c0 + j;
    orow[c] = (yt[row * LSTR + c] - mu) * rs * ln_g[c] + ln_b[c];
  }
}

extern "C" void kernel_launch(void* const* d_in, const int* in_sizes, int n_in,
                              void* d_out, int out_size, void* d_ws, size_t ws_size,
                              hipStream_t stream) {
  const float* embed = (const float*)d_in[0];
  const float* W1    = (const float*)d_in[1];
  const float* b1    = (const float*)d_in[2];
  const float* W2    = (const float*)d_in[3];
  const float* b2    = (const float*)d_in[4];
  const float* Wfc   = (const float*)d_in[5];
  const float* bfc   = (const float*)d_in[6];
  const float* ln_g  = (const float*)d_in[7];
  const float* ln_b  = (const float*)d_in[8];
  float* out = (float*)d_out;

  const size_t BN  = (size_t)BB * NN;        // 16384
  const size_t BNH = BN * HH;                // 8388608

  char* p = (char*)d_ws;
  __bf16* ebf  = (__bf16*)p; p += BNH * 2;
  __bf16* qbf  = (__bf16*)p; p += BNH * 2;   // reused as x2 after argmax
  __bf16* x1bf = (__bf16*)p; p += BNH * 2;
  __bf16* W1t  = (__bf16*)p; p += (size_t)HH * HH * 2;
  __bf16* W2t  = (__bf16*)p; p += (size_t)HH * HH * 2;
  __bf16* Wfct = (__bf16*)p; p += (size_t)2 * HH * HH * 2;
  int*   idx = (int*)p;   p += BN * 4;
  int*   cnt = (int*)p;   p += BN * 4;
  float* g   = (float*)p; p += BN * 4;
  __bf16* x2bf = qbf;

  hipMemsetAsync(cnt, 0, BN * sizeof(int), stream);

  prep_wt_kernel<<<(HH * HH + 255) / 256, 256, 0, stream>>>(W1, W1t, HH, HH);
  prep_wt_kernel<<<(HH * HH + 255) / 256, 256, 0, stream>>>(W2, W2t, HH, HH);
  prep_wt_kernel<<<(2 * HH * HH + 255) / 256, 256, 0, stream>>>(Wfc, Wfct, 2 * HH, HH);

  normalize_kernel<<<BN / 8, 256, 0, stream>>>(embed, ebf, qbf);

  corr_argmax_kernel<<<BB * (NN / 16), 128, 0, stream>>>(qbf, idx, cnt);

  gscale_kernel<<<(BN + 255) / 256, 256, 0, stream>>>(idx, cnt, g);

  // x1 = relu(g .* (embed@W1 + b1))
  gemm_bf16_kernel<<<(BN / 16) * (HH / 16) / 8, 256, 0, stream>>>(
      ebf, W1t, b1, g, x1bf, HH, HH, 1);
  // x2 = g .* (x1@W2 + b2)
  gemm_bf16_kernel<<<(BN / 16) * (HH / 16) / 8, 256, 0, stream>>>(
      x1bf, W2t, b2, g, x2bf, HH, HH, 0);
  // y = LN(concat(embed, x2) @ Wfc + bfc)
  fc_ln_kernel<<<BN / 16, 256, 0, stream>>>(ebf, x2bf, Wfct, bfc, ln_g, ln_b, out);
}
